// CustomSwinTransformer_81587198755382
// MI455X (gfx1250) — matrix operand order repaired and checked by reading, not compiled, not run on hardware
//
#include <hip/hip_runtime.h>
#include <hip/hip_bf16.h>
#include <math.h>

typedef __attribute__((ext_vector_type(16))) _Float16 v16h;
typedef __attribute__((ext_vector_type(8)))  float    v8f;
typedef __attribute__((ext_vector_type(4)))  _Float16 h4;

#define WMMA_F16(a, b, c) \
  __builtin_amdgcn_wmma_f32_16x16x32_f16(false, (a), false, (b), (short)0, (c), false, false)

// ---------------------------------------------------------------------------
// WMMA fragment helpers (wave32).
// A (16x32 f16): lane 0-15 -> M=lane (K 0..7,16..23), lane 16-31 -> M=lane-16
//                (K 8..15,24..31); VGPR v holds K pair kb,kb+1.
//                Per-lane data = two contiguous 8-half runs -> 2x ds_load_b128.
// B (32x16 f16): N = lane&15, K as for A; VGPR v holds K pair kb,kb+1. Stored [K][N] in
//                LDS, so each lane's 16 halves are a column read, ld halves apart.
// C (16x16 f32): VGPR r: lanes 0-15 -> M=r, lanes 16-31 -> M=r+8; N = lane&15.
// ---------------------------------------------------------------------------
static __device__ __forceinline__ v16h ld_frag_A(const _Float16* lds, int row0, int ld) {
  int lane = threadIdx.x & 31;
  int hh = lane >> 4, lr = lane & 15;
  const _Float16* base = lds + (size_t)(row0 + lr) * ld;
  v16h a;
#pragma unroll
  for (int v = 0; v < 8; ++v) {
    int kb = ((v & 4) ? 16 : 0) + hh * 8 + 2 * (v & 3);
    a[2 * v]     = base[kb];
    a[2 * v + 1] = base[kb + 1];
  }
  return a;
}

// LDS holds B K-major as [K][N] rows (k row stride = ld halves, 16B multiple).
static __device__ __forceinline__ v16h ld_frag_B_KN(const _Float16* lds, int k0, int n0, int ld) {
  int lane = threadIdx.x & 31;  // N = lane&15, K in A's table order
  int hh = lane >> 4, lr = lane & 15;
  const _Float16* base = lds + (size_t)k0 * ld + n0 + lr;
  v16h b;
#pragma unroll
  for (int v = 0; v < 8; ++v) {
    int kb = ((v & 4) ? 16 : 0) + hh * 8 + 2 * (v & 3);
    b[2 * v]     = base[(size_t)kb * ld];
    b[2 * v + 1] = base[(size_t)(kb + 1) * ld];
  }
  return b;
}

// ---------------------------------------------------------------------------
// GEMM:  C[M,N] = act(A[M,K] * W[Npad,Kpad]^T + bias)
// Single straight-line path: W is pre-padded (Npad mult of 64, Kpad mult of
// 32, zero-filled), A lives in workspace so tile over-reads are safe and the
// K-tail garbage multiplies W's zero padding. No staging guards at all.
// Block: 256 threads = 8 waves; tile 128(M) x 64(N) x 32(K); wave = 32x32.
// ---------------------------------------------------------------------------
#define G_TM 128
#define G_TN 64
#define G_TK 32
#define G_LDA 40   // halves; 80B row stride (16B multiple)
#define G_LDWT 72  // halves; 144B row stride (16B multiple)

__global__ __launch_bounds__(256) void gemm_f16wmma_kernel(
    const float* __restrict__ A, const float* __restrict__ W,
    const float* __restrict__ bias, float* __restrict__ C,
    int M, int N, int K, int Kpad, int act) {
  __shared__ _Float16 sA[G_TM * G_LDA];    // [m][k]
  __shared__ _Float16 sWt[G_TK * G_LDWT];  // [k][n] (transposed while staging)

  int m0 = blockIdx.x * G_TM;
  int n0 = blockIdx.y * G_TN;
  int t = threadIdx.x;
  int wid = t >> 5;
  int wm = wid >> 1, wn = wid & 1;  // 4x2 wave grid

  v8f acc[2][2] = {{{0}}, {{0}}};

  for (int k0 = 0; k0 < Kpad; k0 += G_TK) {
    // A tile: 128x32 floats = 1024 float4; 4 per thread (unguarded).
#pragma unroll
    for (int i = 0; i < 4; ++i) {
      int q = t + i * 256;
      int r = q >> 3, c4 = (q & 7) * 4;
      const float4 v = *(const float4*)(A + (size_t)(m0 + r) * K + k0 + c4);
      h4 hv = {(_Float16)v.x, (_Float16)v.y, (_Float16)v.z, (_Float16)v.w};
      *(h4*)(sA + r * G_LDA + c4) = hv;
    }
    // W tile: 64x32 floats = 512 float4; 2 per thread; store transposed.
#pragma unroll
    for (int i = 0; i < 2; ++i) {
      int q = t + i * 256;
      int r = q >> 3, c4 = (q & 7) * 4;
      const float4 v = *(const float4*)(W + (size_t)(n0 + r) * Kpad + k0 + c4);
      sWt[(c4 + 0) * G_LDWT + r] = (_Float16)v.x;
      sWt[(c4 + 1) * G_LDWT + r] = (_Float16)v.y;
      sWt[(c4 + 2) * G_LDWT + r] = (_Float16)v.z;
      sWt[(c4 + 3) * G_LDWT + r] = (_Float16)v.w;
    }
    __syncthreads();

    v16h af[2], bf[2];
    af[0] = ld_frag_A(sA, wm * 32, G_LDA);
    af[1] = ld_frag_A(sA, wm * 32 + 16, G_LDA);
    bf[0] = ld_frag_B_KN(sWt, 0, wn * 32, G_LDWT);
    bf[1] = ld_frag_B_KN(sWt, 0, wn * 32 + 16, G_LDWT);
#pragma unroll
    for (int mi = 0; mi < 2; ++mi)
#pragma unroll
      for (int ni = 0; ni < 2; ++ni)
        acc[mi][ni] = WMMA_F16(af[mi], bf[ni], acc[mi][ni]);
    __syncthreads();
  }

  int lane = t & 31;
  int hh = lane >> 4, lr = lane & 15;
#pragma unroll
  for (int mi = 0; mi < 2; ++mi)
#pragma unroll
    for (int ni = 0; ni < 2; ++ni) {
      int n = n0 + wn * 32 + ni * 16 + lr;
      int nc = (n < N) ? n : (N - 1);
      float bv = (bias != nullptr) ? bias[nc] : 0.f;
#pragma unroll
      for (int r = 0; r < 8; ++r) {
        int m = m0 + wm * 32 + mi * 16 + r + hh * 8;
        if (m < M && n < N) {
          float xv = acc[mi][ni][r] + bv;
          if (act == 1)  // exact GELU
            xv = 0.5f * xv * (1.f + erff(xv * 0.70710678118654752f));
          C[(size_t)m * N + n] = xv;
        }
      }
    }
}

// Zero-pad a weight matrix (N,K) row-major into (Npad,Kpad).
__global__ void pad_weight_kernel(const float* __restrict__ W,
                                  float* __restrict__ Wp, int N, int K,
                                  int Npad, int Kpad) {
  size_t total = (size_t)Npad * Kpad;
  for (size_t i = (size_t)blockIdx.x * blockDim.x + threadIdx.x; i < total;
       i += (size_t)gridDim.x * blockDim.x) {
    int k = (int)(i % Kpad);
    int n = (int)(i / Kpad);
    int cn = (n < N) ? n : (N - 1);
    int ck = (k < K) ? k : (K - 1);
    float v = W[(size_t)cn * K + ck];
    Wp[i] = (n < N && k < K) ? v : 0.f;
  }
}

// ---------------------------------------------------------------------------
// Window attention core. One block (128 thr = 4 waves) per (window, head).
// qkv: (nW*64, 3C); o: (nW*64, C). d = C/nh = 32 always, N = 64 tokens.
// ---------------------------------------------------------------------------
__global__ __launch_bounds__(128) void attn_wmma_kernel(
    const float* __restrict__ qkv, float* __restrict__ o,
    const float* __restrict__ bias_tab,     // (nh,64,64)
    const float* __restrict__ logit_scale,  // (nh)
    int nh, int C, int nww, int shifted) {
  const int d = 32;
  __shared__ _Float16 sQ[64 * 40];   // [token][k], 80B rows
  __shared__ _Float16 sKT[32 * 72];  // [k][token], 144B rows (B for S)
  __shared__ _Float16 sV[64 * 40];   // [token=K][ch=N], 80B rows (B for O)
  __shared__ float    sS[64 * 68];
  __shared__ _Float16 sP[64 * 72];   // [row][m], 144B rows (A for O)
  __shared__ int      sG[64];
  __shared__ float    sScale;

  int wIdx = blockIdx.x;
  int h = blockIdx.y;
  int t = threadIdx.x;
  size_t rowbase = (size_t)wIdx * 64;
  int C3 = 3 * C;

  if (t == 0) sScale = expf(fminf(logit_scale[h], 4.6051701859880914f));  // ln(100)

  if (t < 64) {
    const float* qr = qkv + (rowbase + t) * (size_t)C3 + h * d;
    const float* kr = qr + C;
    const float* vr = qr + 2 * C;
    float4 qv[8], kv[8], vv[8];
    float qs = 0.f, ks = 0.f;
#pragma unroll
    for (int i = 0; i < 8; ++i) {
      qv[i] = *(const float4*)(qr + 4 * i);
      kv[i] = *(const float4*)(kr + 4 * i);
      vv[i] = *(const float4*)(vr + 4 * i);
      qs += qv[i].x * qv[i].x + qv[i].y * qv[i].y + qv[i].z * qv[i].z + qv[i].w * qv[i].w;
      ks += kv[i].x * kv[i].x + kv[i].y * kv[i].y + kv[i].z * kv[i].z + kv[i].w * kv[i].w;
    }
    float qi = 1.f / fmaxf(sqrtf(qs), 1e-12f);
    float ki = 1.f / fmaxf(sqrtf(ks), 1e-12f);
#pragma unroll
    for (int i = 0; i < 8; ++i) {
      h4 qh = {(_Float16)(qv[i].x * qi), (_Float16)(qv[i].y * qi),
               (_Float16)(qv[i].z * qi), (_Float16)(qv[i].w * qi)};
      *(h4*)(sQ + t * 40 + 4 * i) = qh;
      h4 vh = {(_Float16)vv[i].x, (_Float16)vv[i].y, (_Float16)vv[i].z, (_Float16)vv[i].w};
      *(h4*)(sV + t * 40 + 4 * i) = vh;
      sKT[(4 * i + 0) * 72 + t] = (_Float16)(kv[i].x * ki);
      sKT[(4 * i + 1) * 72 + t] = (_Float16)(kv[i].y * ki);
      sKT[(4 * i + 2) * 72 + t] = (_Float16)(kv[i].z * ki);
      sKT[(4 * i + 3) * 72 + t] = (_Float16)(kv[i].w * ki);
    }
    if (shifted) {
      int H = nww * 8;
      int wi = wIdx % (nww * nww);
      int py = (wi / nww) * 8 + (t >> 3);
      int px = (wi % nww) * 8 + (t & 7);
      int gy = (py < H - 8) ? 0 : ((py < H - 4) ? 1 : 2);
      int gx = (px < H - 8) ? 0 : ((px < H - 4) ? 1 : 2);
      sG[t] = gy * 3 + gx;
    } else {
      sG[t] = 0;
    }
  }
  __syncthreads();

  // S = qn * kn^T : each wave does a 16-row band, K = 32 in one WMMA step.
  int wv = t >> 5;
  {
    v16h a = ld_frag_A(sQ, wv * 16, 40);
    int lane = t & 31, hh = lane >> 4, lr = lane & 15;
#pragma unroll
    for (int nt = 0; nt < 4; ++nt) {
      v16h b = ld_frag_B_KN(sKT, 0, nt * 16, 72);
      v8f c = {0};
      c = WMMA_F16(a, b, c);
#pragma unroll
      for (int r = 0; r < 8; ++r)
        sS[(wv * 16 + r + hh * 8) * 68 + nt * 16 + lr] = c[r];
    }
  }
  __syncthreads();

  // softmax rows (logit scale, cpb bias, shift mask folded here)
  if (t < 64) {
    float scale = sScale;
    const float* brow = bias_tab + ((size_t)h * 64 + t) * 64;
    int gn = sG[t];
    float mx = -1e30f;
    for (int m = 0; m < 64; ++m) {
      float x = sS[t * 68 + m] * scale + brow[m] + ((sG[m] != gn) ? -100.f : 0.f);
      mx = fmaxf(mx, x);
    }
    float sum = 0.f;
    for (int m = 0; m < 64; ++m) {
      float x = sS[t * 68 + m] * scale + brow[m] + ((sG[m] != gn) ? -100.f : 0.f);
      float e = expf(x - mx);
      sum += e;
      sP[t * 72 + m] = (_Float16)e;
    }
    float inv = 1.f / sum;
    for (int m = 0; m < 64; ++m)
      sP[t * 72 + m] = (_Float16)((float)sP[t * 72 + m] * inv);
  }
  __syncthreads();

  // O = P * V : K = 64 in two steps, N = 32 in two tiles.
  v8f oc[2] = {{0}, {0}};
#pragma unroll
  for (int ks = 0; ks < 2; ++ks) {
    v16h a = ld_frag_A(sP + ks * 32, wv * 16, 72);
#pragma unroll
    for (int nt = 0; nt < 2; ++nt) {
      v16h b = ld_frag_B_KN(sV, ks * 32, nt * 16, 40);
      oc[nt] = WMMA_F16(a, b, oc[nt]);
    }
  }
  int lane = t & 31, hh = lane >> 4, lr = lane & 15;
#pragma unroll
  for (int nt = 0; nt < 2; ++nt)
#pragma unroll
    for (int r = 0; r < 8; ++r) {
      int row = wv * 16 + r + hh * 8;
      o[(rowbase + row) * (size_t)C + h * d + nt * 16 + lr] = oc[nt][r];
    }
}

// ---------------------------------------------------------------------------
// Continuous position bias: regenerate REL_TABLE / REL_IDX on device,
// tbl = relu(T @ w1^T + b1) @ w2^T ; bias[h,n,m] = 16*sigmoid(tbl[idx(n,m),h])
// One block per head.
// ---------------------------------------------------------------------------
__global__ __launch_bounds__(256) void cpb_bias_kernel(
    const float* __restrict__ w1, const float* __restrict__ b1,
    const float* __restrict__ w2, float* __restrict__ bias_tab, int nh) {
  __shared__ float tbl[225];
  int h = blockIdx.x;
  int t = threadIdx.x;
  if (t < 225) {
    int ry = t / 15 - 7, rx = t % 15 - 7;
    float ty = (float)ry * (8.f / 7.f);
    float tx = (float)rx * (8.f / 7.f);
    float sy = (ty > 0.f) ? 1.f : ((ty < 0.f) ? -1.f : 0.f);
    float sx = (tx > 0.f) ? 1.f : ((tx < 0.f) ? -1.f : 0.f);
    float t0 = sy * log2f(fabsf(ty) + 1.f) * (1.f / 3.f);
    float t1 = sx * log2f(fabsf(tx) + 1.f) * (1.f / 3.f);
    float acc = 0.f;
    const float* w2h = w2 + (size_t)h * 512;
    for (int j = 0; j < 512; ++j) {
      float hd = w1[2 * j] * t0 + w1[2 * j + 1] * t1 + b1[j];
      hd = fmaxf(hd, 0.f);
      acc += hd * w2h[j];
    }
    tbl[t] = acc;
  }
  __syncthreads();
  for (int i = t; i < 4096; i += 256) {
    int n = i >> 6, m = i & 63;
    int dy = (n >> 3) - (m >> 3) + 7;
    int dx = (n & 7) - (m & 7) + 7;
    float v = tbl[dy * 15 + dx];
    bias_tab[(size_t)h * 4096 + i] = 16.f / (1.f + expf(-v));
  }
}

// ---------------------------------------------------------------------------
// LayerNorm over last dim. mode 0: dst = LN(src)*g+b ; mode 1: dst += LN(src).
// One wave (32 lanes) per row; 8 rows per 256-thread block.
// ---------------------------------------------------------------------------
__global__ __launch_bounds__(256) void ln_kernel(
    float* __restrict__ dst, const float* __restrict__ src,
    const float* __restrict__ g, const float* __restrict__ b,
    size_t rows, int C, int mode) {
  int lane = threadIdx.x & 31;
  int wid = threadIdx.x >> 5;
  size_t row = (size_t)blockIdx.x * 8 + wid;
  if (row >= rows) return;
  const float* sr = src + row * C;
  float s = 0.f;
  for (int i = lane; i < C; i += 32) s += sr[i];
#pragma unroll
  for (int off = 16; off; off >>= 1) s += __shfl_xor(s, off, 32);
  float mean = s / (float)C;
  float v = 0.f;
  for (int i = lane; i < C; i += 32) { float d = sr[i] - mean; v += d * d; }
#pragma unroll
  for (int off = 16; off; off >>= 1) v += __shfl_xor(v, off, 32);
  float inv = rsqrtf(v / (float)C + 1e-5f);
  float* dr = dst + row * C;
  for (int i = lane; i < C; i += 32) {
    float o = (sr[i] - mean) * inv * g[i] + b[i];
    dr[i] = mode ? (dr[i] + o) : o;
  }
}

// ---------------------------------------------------------------------------
// Data-movement glue kernels
// ---------------------------------------------------------------------------
#define GSTRIDE(i, total)                                                     \
  for (size_t i = (size_t)blockIdx.x * blockDim.x + threadIdx.x; i < (total); \
       i += (size_t)gridDim.x * blockDim.x)

__global__ void patch_rearrange_kernel(const float* __restrict__ x,
                                       float* __restrict__ xp, int Bn) {
  size_t total = (size_t)Bn * 64 * 64 * 48;
  GSTRIDE(i, total) {
    int cc = (int)(i % 48); size_t r = i / 48;
    int gx = (int)(r % 64); size_t r2 = r / 64;
    int gy = (int)(r2 % 64); int b = (int)(r2 / 64);
    int c = cc >> 4, py = (cc >> 2) & 3, px = cc & 3;
    xp[i] = x[(((size_t)b * 3 + c) * 256 + gy * 4 + py) * 256 + gx * 4 + px];
  }
}

__global__ void window_partition_kernel(const float* __restrict__ x,
                                        float* __restrict__ xw, int Bn, int H,
                                        int C, int sh) {
  size_t total = (size_t)Bn * H * H * C;
  int nww = H >> 3;
  GSTRIDE(i, total) {
    int c = (int)(i % C); size_t r = i / C;
    int pos = (int)(r % 64); size_t wr = r / 64;
    int wx = (int)(wr % nww); size_t t2 = wr / nww;
    int wy = (int)(t2 % nww); int b = (int)(t2 / nww);
    int sy = (wy * 8 + (pos >> 3) + sh) % H;
    int sx = (wx * 8 + (pos & 7) + sh) % H;
    xw[i] = x[(((size_t)b * H + sy) * H + sx) * C + c];
  }
}

__global__ void window_reverse_kernel(const float* __restrict__ xw,
                                      float* __restrict__ x, int Bn, int H,
                                      int C, int sh) {
  size_t total = (size_t)Bn * H * H * C;
  int nww = H >> 3;
  GSTRIDE(i, total) {
    int c = (int)(i % C); size_t r = i / C;
    int px = (int)(r % H); size_t r2 = r / H;
    int py = (int)(r2 % H); int b = (int)(r2 / H);
    int ry = (py - sh + H) % H, rx = (px - sh + H) % H;
    size_t tok = (((size_t)b * nww + (ry >> 3)) * nww + (rx >> 3)) * 64 +
                 (ry & 7) * 8 + (rx & 7);
    x[i] = xw[tok * C + c];
  }
}

__global__ void merge_cat_kernel(const float* __restrict__ x,
                                 float* __restrict__ cat, int Bn, int H, int C) {
  int H2 = H >> 1;
  size_t total = (size_t)Bn * H2 * H2 * 4 * C;
  GSTRIDE(i, total) {
    int ch = (int)(i % (4 * C)); size_t r = i / (4 * C);
    int q = ch / C, c = ch % C;
    int ox = (int)(r % H2); size_t r2 = r / H2;
    int oy = (int)(r2 % H2); int b = (int)(r2 / H2);
    int dy = q & 1, dx = q >> 1;
    cat[i] = x[(((size_t)b * H + 2 * oy + dy) * H + 2 * ox + dx) * C + c];
  }
}

__global__ void meanpool_kernel(const float* __restrict__ x,
                                float* __restrict__ feat, int C, int HW) {
  int b = blockIdx.x;
  for (int c = threadIdx.x; c < C; c += blockDim.x) {
    float s = 0.f;
    for (int p = 0; p < HW; ++p) s += x[((size_t)b * HW + p) * C + c];
    feat[(size_t)b * C + c] = s / (float)HW;
  }
}

__global__ void reparam_kernel(const float* __restrict__ stat,
                               const float* __restrict__ eps,
                               float* __restrict__ z, float* __restrict__ mu_o,
                               float* __restrict__ std_o) {
  int i = blockIdx.x * blockDim.x + threadIdx.x;
  if (i >= 16 * 512) return;
  int b = i >> 9, j = i & 511;
  float mu = stat[b * 1024 + j];
  float s = stat[b * 1024 + 512 + j] - 5.f;
  float sp = (s > 20.f) ? s : log1pf(expf(s));
  z[i] = mu + sp * eps[i];
  mu_o[i] = mu;
  std_o[i] = sp;
}

// ---------------------------------------------------------------------------
// Host orchestration
// ---------------------------------------------------------------------------
static inline size_t cdiv(size_t a, size_t b) { return (a + b - 1) / b; }

extern "C" void kernel_launch(void* const* d_in, const int* in_sizes, int n_in,
                              void* d_out, int out_size, void* d_ws,
                              size_t ws_size, hipStream_t stream) {
  (void)in_sizes; (void)n_in; (void)out_size; (void)ws_size;
  static const int DEP[4] = {2, 2, 6, 2};
  static const int NH[4]  = {3, 6, 12, 24};
  const int Bn = 16;

  const float** in = (const float**)d_in;
  int idx = 0;
  const float* x_in    = in[idx++];
  const float* patch_w = in[idx++];
  const float* patch_b = in[idx++];
  const float* pe_g    = in[idx++];
  const float* pe_b    = in[idx++];
  struct Blk { const float* p[16]; };
  Blk blk[12];
  const float* mrg[3][3];
  int bc = 0;
  for (int si = 0; si < 4; ++si) {
    for (int d = 0; d < DEP[si]; ++d) {
      for (int j = 0; j < 16; ++j) blk[bc].p[j] = in[idx++];
      ++bc;
    }
    if (si < 3) for (int j = 0; j < 3; ++j) mrg[si][j] = in[idx++];
  }
  const float* fng = in[idx++];
  const float* fnb = in[idx++];
  const float* h1w = in[idx++];
  const float* h1b = in[idx++];
  const float* h2w = in[idx++];
  const float* h2b = in[idx++];
  const float* eps = in[idx++];

  // workspace carve-up (floats)
  float* ws = (float*)d_ws;
  size_t off = 0;
  auto alloc = [&](size_t n) { float* p = ws + off; off += n; return p; };
  const size_t MAXTC = (size_t)Bn * 64 * 64 * 96;  // 6.29M floats
  float* X    = alloc(MAXTC);
  float* Y    = alloc(MAXTC);
  float* WINb = alloc(MAXTC);
  float* O    = alloc(MAXTC);
  float* BIG  = alloc(MAXTC * 4);  // qkv (3C) & MLP hidden (4C)
  float* BIAS = alloc((size_t)24 * 4096);
  float* FEAT = alloc(16 * 768);
  float* STAT = alloc(16 * 1024);
  float* Zb   = alloc(16 * 512);
  float* WPAD = alloc((size_t)8640 * 512);  // largest padded weight (head2)
  (void)alloc((size_t)G_TM * 3072);         // slack for A-tile over-reads

  auto gemm = [&](const float* A, const float* W, const float* bias, float* C,
                  size_t M, int N, int K, int act) {
    int Npad = (int)cdiv((size_t)N, G_TN) * G_TN;
    int Kpad = (int)cdiv((size_t)K, G_TK) * G_TK;
    const float* Wuse = W;
    if (Npad != N || Kpad != K) {
      size_t tot = (size_t)Npad * Kpad;
      pad_weight_kernel<<<(unsigned)cdiv(tot, 256), 256, 0, stream>>>(
          W, WPAD, N, K, Npad, Kpad);
      Wuse = WPAD;
    }
    dim3 g((unsigned)cdiv(M, G_TM), (unsigned)(Npad / G_TN));
    gemm_f16wmma_kernel<<<g, 256, 0, stream>>>(A, Wuse, bias, C, (int)M, N, K,
                                               Kpad, act);
  };
  auto copyk = [&](size_t total) { return (unsigned)cdiv(total, 256); };

  // ---- patch embed ----
  size_t ntok = (size_t)Bn * 64 * 64;
  patch_rearrange_kernel<<<copyk(ntok * 48), 256, 0, stream>>>(x_in, WINb, Bn);
  gemm(WINb, patch_w, patch_b, Y, ntok, 96, 48, 0);
  ln_kernel<<<(unsigned)cdiv(ntok, 8), 256, 0, stream>>>(X, Y, pe_g, pe_b, ntok, 96, 0);

  int H = 64, C = 96;
  bc = 0;
  for (int si = 0; si < 4; ++si) {
    int nh = NH[si];
    ntok = (size_t)Bn * H * H;
    int nww = H >> 3;
    int nW = Bn * nww * nww;
    for (int d = 0; d < DEP[si]; ++d, ++bc) {
      const Blk& bp = blk[bc];
      int sh = ((d & 1) && H > 8) ? 4 : 0;
      // window partition (roll fused)
      window_partition_kernel<<<copyk(ntok * C), 256, 0, stream>>>(X, WINb, Bn, H, C, sh);
      // qkv projection
      gemm(WINb, bp.p[2], bp.p[3], BIG, ntok, 3 * C, C, 0);
      // continuous position bias table
      cpb_bias_kernel<<<nh, 256, 0, stream>>>(bp.p[5], bp.p[6], bp.p[7], BIAS, nh);
      // attention core
      dim3 ag((unsigned)nW, (unsigned)nh);
      attn_wmma_kernel<<<ag, 128, 0, stream>>>(BIG, O, BIAS, bp.p[4], nh, C, nww, sh ? 1 : 0);
      // output projection
      gemm(O, bp.p[8], bp.p[9], WINb, ntok, C, C, 0);
      // window reverse (roll back fused)
      window_reverse_kernel<<<copyk(ntok * C), 256, 0, stream>>>(WINb, Y, Bn, H, C, sh);
      // x += LN(attn)
      ln_kernel<<<(unsigned)cdiv(ntok, 8), 256, 0, stream>>>(X, Y, bp.p[0], bp.p[1], ntok, C, 1);
      // MLP
      gemm(X, bp.p[12], bp.p[13], BIG, ntok, 4 * C, C, 1);   // fc1 + GELU
      gemm(BIG, bp.p[14], bp.p[15], O, ntok, C, 4 * C, 0);   // fc2
      ln_kernel<<<(unsigned)cdiv(ntok, 8), 256, 0, stream>>>(X, O, bp.p[10], bp.p[11], ntok, C, 1);
    }
    if (si < 3) {
      int H2 = H >> 1;
      size_t ntok2 = (size_t)Bn * H2 * H2;
      merge_cat_kernel<<<copyk(ntok2 * 4 * C), 256, 0, stream>>>(X, WINb, Bn, H, C);
      gemm(WINb, mrg[si][0], nullptr, Y, ntok2, 2 * C, 4 * C, 0);
      ln_kernel<<<(unsigned)cdiv(ntok2, 8), 256, 0, stream>>>(X, Y, mrg[si][1], mrg[si][2], ntok2, 2 * C, 0);
      H = H2;
      C *= 2;
    }
  }

  // ---- head ----
  ntok = (size_t)Bn * H * H;  // 16*64 tokens, C=768
  ln_kernel<<<(unsigned)cdiv(ntok, 8), 256, 0, stream>>>(Y, X, fng, fnb, ntok, C, 0);
  meanpool_kernel<<<Bn, 256, 0, stream>>>(Y, FEAT, C, H * H);
  gemm(FEAT, h1w, h1b, STAT, 16, 1024, 768, 0);

  float* outp  = (float*)d_out;
  float* mu_o  = outp + (size_t)16 * 8631;
  float* std_o = mu_o + (size_t)16 * 512;
  reparam_kernel<<<32, 256, 0, stream>>>(STAT, eps, Zb, mu_o, std_o);
  gemm(Zb, h2w, h2b, outp, 16, 8631, 512, 0);
}
